// Self_Attention_23278722744806
// MI455X (gfx1250) — matrix operand order repaired and checked by reading, not compiled, not run on hardware
//
#include <hip/hip_runtime.h>

// ---------------------------------------------------------------------------
// Self-attention, B=4 S=2048 D=1024, via f16 WMMA GEMMs with f32 accumulate.
// Pipeline: cvt(f32->f16) -> Q,K proj -> V proj (transposed) ->
//           per batch: scores (Q K^T * NORM) -> row softmax -> P @ Vt^T.
// All GEMMs are C = A * Bsrc^T with A[M,K], Bsrc[N,K] row-major, which makes
// every CDNA5 WMMA fragment a contiguous per-lane load (no transpose insts).
// Per-wave tile: 32x64 (2 A-frags x 4 B-frags -> 8 WMMAs per 32-K step),
// so each B fragment load feeds two v_wmma and the load:WMMA ratio is 1.5.
// ---------------------------------------------------------------------------

typedef __attribute__((ext_vector_type(16))) _Float16 v16h;
typedef __attribute__((ext_vector_type(8)))  _Float16 v8h;
typedef __attribute__((ext_vector_type(8)))  float    v8f;

__global__ void sa_f32_to_f16(const float* __restrict__ src,
                              _Float16* __restrict__ dst, int n) {
    int i = blockIdx.x * blockDim.x + threadIdx.x;
    if (i < n) dst[i] = (_Float16)src[i];
}

// OUT_MODE: 0 = f32 (scaled), 1 = f16, 2 = f16 transposed (C[n*ldc + m])
template <int OUT_MODE>
__global__ __launch_bounds__(128) void sa_gemm_nt(
    const _Float16* __restrict__ A,  int lda,   // A   [M, K] row-major
    const _Float16* __restrict__ Bm, int ldb,   // Bm  [N, K] row-major
    void* __restrict__ Cout, int ldc,
    int K, float scale)
{
    const int lane    = threadIdx.x & 31;
    const int wave    = threadIdx.x >> 5;
    const int half_hi = lane >> 4;       // 0: lanes 0-15, 1: lanes 16-31
    const int lr      = lane & 15;

    const int m0 = (blockIdx.x * 4 + wave) * 32;   // 32 rows per wave
    const int n0 = blockIdx.y * 64;                // 4 x 16-col tiles per wave

    // A fragment (16x32 f16): lane holds row m+lr; K chunks at
    // kb..kb+7 and kb+16..kb+23 where kb = 8*half_hi.
    const _Float16* arow0 = A + (size_t)(m0 + lr) * lda + half_hi * 8;
    const _Float16* arow1 = arow0 + (size_t)16 * lda;
    // B fragment (32x16 f16): lane holds column n0+lr; K chunks at
    // kb..kb+7 and kb+16..kb+23 where kb = 8*half_hi.
    const _Float16* brow = Bm + (size_t)(n0 + lr) * ldb + half_hi * 8;
    const size_t bstep = (size_t)16 * ldb;

    v8f acc[2][4] = {};

    union AFrag { v16h v; v8h h[2]; };

    for (int k = 0; k < K; k += 32) {
        AFrag a0, a1;
        a0.h[0] = *(const v8h*)(arow0 + k);
        a0.h[1] = *(const v8h*)(arow0 + k + 16);
        a1.h[0] = *(const v8h*)(arow1 + k);
        a1.h[1] = *(const v8h*)(arow1 + k + 16);
        __builtin_prefetch(arow0 + k + 128, 0, 0);
        __builtin_prefetch(arow1 + k + 128, 0, 0);
        __builtin_prefetch(brow + k + 128, 0, 0);
#pragma unroll
        for (int j = 0; j < 4; ++j) {
            AFrag bf; bf.h[0] = *(const v8h*)(brow + (size_t)j * bstep + k); bf.h[1] = *(const v8h*)(brow + (size_t)j * bstep + k + 16); v16h b = bf.v;
            acc[0][j] = __builtin_amdgcn_wmma_f32_16x16x32_f16(
                false, a0.v, false, b, (short)0, acc[0][j], false, false);
            acc[1][j] = __builtin_amdgcn_wmma_f32_16x16x32_f16(
                false, a1.v, false, b, (short)0, acc[1][j], false, false);
        }
    }

    // C/D layout: VGPR r holds (M = r + 8*half_hi, N = lr).
#pragma unroll
    for (int g = 0; g < 2; ++g) {
#pragma unroll
        for (int j = 0; j < 4; ++j) {
#pragma unroll
            for (int r = 0; r < 8; ++r) {
                const int m = m0 + g * 16 + r + 8 * half_hi;
                const int n = n0 + j * 16 + lr;
                const float v = acc[g][j][r] * scale;
                if (OUT_MODE == 0) {
                    ((float*)Cout)[(size_t)m * ldc + n] = v;
                } else if (OUT_MODE == 1) {
                    ((_Float16*)Cout)[(size_t)m * ldc + n] = (_Float16)v;
                } else {
                    ((_Float16*)Cout)[(size_t)n * ldc + m] = (_Float16)v;
                }
            }
        }
    }
}

// One wave per row: max, sum(exp), normalize -> f16 probabilities.
__global__ __launch_bounds__(128) void sa_softmax_rows(
    const float* __restrict__ Sc, _Float16* __restrict__ P, int S)
{
    const int row  = blockIdx.x * 4 + (threadIdx.x >> 5);
    const int lane = threadIdx.x & 31;
    const float* r = Sc + (size_t)row * S;

    float m = -__builtin_inff();
    for (int i = lane; i < S; i += 32) m = fmaxf(m, r[i]);
#pragma unroll
    for (int off = 16; off > 0; off >>= 1)
        m = fmaxf(m, __shfl_xor(m, off, 32));

    float sum = 0.0f;
    for (int i = lane; i < S; i += 32) sum += __expf(r[i] - m);
#pragma unroll
    for (int off = 16; off > 0; off >>= 1)
        sum += __shfl_xor(sum, off, 32);

    const float inv = 1.0f / sum;
    _Float16* p = P + (size_t)row * S;
    for (int i = lane; i < S; i += 32)
        p[i] = (_Float16)(__expf(r[i] - m) * inv);
}

extern "C" void kernel_launch(void* const* d_in, const int* in_sizes, int n_in,
                              void* d_out, int out_size, void* d_ws, size_t ws_size,
                              hipStream_t stream) {
    (void)in_sizes; (void)n_in; (void)out_size; (void)ws_size;
    constexpr int B = 4, S = 2048, D = 1024;
    constexpr float NORM = 0.03125f;  // 1/sqrt(1024)
    const size_t BS = (size_t)B * S;

    const float* x  = (const float*)d_in[0];
    const float* Wq = (const float*)d_in[1];
    const float* Wk = (const float*)d_in[2];
    const float* Wv = (const float*)d_in[3];
    float* out = (float*)d_out;

    // Workspace layout (all regions 256B aligned):
    char* ws = (char*)d_ws;
    _Float16* xh  = (_Float16*)(ws);                                  // 16 MB
    _Float16* Wqh = (_Float16*)(ws + (16u << 20));                    //  2 MB
    _Float16* Wkh = (_Float16*)(ws + (16u << 20) + (2u << 20));       //  2 MB
    _Float16* Wvh = (_Float16*)(ws + (16u << 20) + (4u << 20));       //  2 MB
    _Float16* Qh  = (_Float16*)(ws + (22u << 20));                    // 16 MB
    _Float16* Kh  = Qh + BS * D;                                      // 16 MB
    _Float16* Vt  = Kh + BS * D;   // [b][e][s] transposed V          // 16 MB
    _Float16* P   = Vt + BS * D;   // per-batch probs [S,S] f16       //  8 MB
    float*    Sc  = (float*)xh;    // per-batch scores f32, aliases xh (dead)

    // 1) fp32 -> f16 conversions
    sa_f32_to_f16<<<(int)((BS * D + 255) / 256), 256, 0, stream>>>(x, xh, (int)(BS * D));
    sa_f32_to_f16<<<(D * D + 255) / 256, 256, 0, stream>>>(Wq, Wqh, D * D);
    sa_f32_to_f16<<<(D * D + 255) / 256, 256, 0, stream>>>(Wk, Wkh, D * D);
    sa_f32_to_f16<<<(D * D + 255) / 256, 256, 0, stream>>>(Wv, Wvh, D * D);

    // 2) Q/K projections: [BS,D] = xh[BS,D] @ W[D,D]^T  (f16 out)
    dim3 gqk((unsigned)(BS / 128), D / 64);
    sa_gemm_nt<1><<<gqk, 128, 0, stream>>>(xh, D, Wqh, D, (void*)Qh, D, D, 1.0f);
    sa_gemm_nt<1><<<gqk, 128, 0, stream>>>(xh, D, Wkh, D, (void*)Kh, D, D, 1.0f);

    // 3) V projection, written transposed per batch: Vt[b][e][s]
    dim3 gv(S / 128, D / 64);
    for (int b = 0; b < B; ++b) {
        sa_gemm_nt<2><<<gv, 128, 0, stream>>>(
            xh + (size_t)b * S * D, D, Wvh, D,
            (void*)(Vt + (size_t)b * D * S), S, D, 1.0f);
    }

    // 4) per-batch attention (score buffer reused; stream order = dependency)
    dim3 gsc(S / 128, S / 64);
    dim3 gpv(S / 128, D / 64);
    for (int b = 0; b < B; ++b) {
        // scores = (Q K^T) * NORM  -> f32 [S,S]
        sa_gemm_nt<0><<<gsc, 128, 0, stream>>>(
            Qh + (size_t)b * S * D, D, Kh + (size_t)b * S * D, D,
            (void*)Sc, S, D, NORM);
        // softmax rows -> P f16
        sa_softmax_rows<<<S / 4, 128, 0, stream>>>(Sc, P, S);
        // out[b] = P @ Vt[b]^T  (K = S), f32 out
        sa_gemm_nt<0><<<gpv, 128, 0, stream>>>(
            P, S, Vt + (size_t)b * D * S, S,
            (void*)(out + (size_t)b * S * D), D, S, 1.0f);
    }
}
